// S4DLayer_12008728559951
// MI455X (gfx1250) — matrix operand order repaired and checked by reading, not compiled, not run on hardware
//
#include <hip/hip_runtime.h>

// ---------------------------------------------------------------------------
// S4D layer via chunked scan, CDNA5 (gfx1250) WMMA f16 16x16x32, wave32.
//   y[b,h,:] = causal_conv(u[b,h,:], K[h,:]) + D[h]*u[b,h,:]
// Chunk size C=64, NQ=64 chunks. Per (b,h):
//   P = Win @ U        (complex in x real)      -> WMMA x2
//   s[q+1] = w^64 s[q] + P[:,q]                 -> tiny f32 scan
//   Y = T @ U + Re(Wout) @ Sre + (-Im(Wout)) @ Sim   -> WMMA x3 (accumulated)
// ---------------------------------------------------------------------------

typedef __attribute__((ext_vector_type(16))) _Float16 v16h;
typedef __attribute__((ext_vector_type(8)))  _Float16 v8h;
typedef __attribute__((ext_vector_type(8)))  float    v8f;

#define BDIM 16
#define HDIM 256
#define LDIM 4096
#define NDIM 64

// per-h workspace layout (bytes)
#define OFF_T       0        // 64x64 f16: Toeplitz kernel tile (+D on diag)
#define OFF_WINRE   8192     // 64x64 f16: Re(w^{63-j}),  rows n, cols j
#define OFF_WINIM   16384    // 64x64 f16: Im(w^{63-j})
#define OFF_WOUTRE  24576    // 64x64 f16: Re(coef*w^{i+1}), rows i, cols n
#define OFF_WOUTIMN 32768    // 64x64 f16: -Im(coef*w^{i+1})
#define OFF_WC      40960    // 64 f32 re, 64 f32 im of w^64
#define WS_STRIDE   41472    // per-h stride (multiple of 128)

static __device__ inline v16h cat16(v8h lo, v8h hi) {
    union { v16h v; v8h h[2]; } u;
    u.h[0] = lo; u.h[1] = hi;
    return u.v;
}

// A operand (M x K = 16x32 f16), ISA 7.12.2 layout:
// lane L: m = mt*16 + (L&15); half-wave selects K offset 0/8; VGPR v<4 -> K=koff+2v,
// v>=4 -> K=16+koff+2(v-4). Two contiguous 8-half (16B) segments per lane.
static __device__ inline v16h load_a64(const _Float16* __restrict__ mat,
                                       int lane, int mt, int kb) {
    int m     = mt * 16 + (lane & 15);
    int koff  = (lane >> 4) ? 8 : 0;
    int kbase = kb * 32 + koff;
    const v8h* p0 = reinterpret_cast<const v8h*>(mat + m * 64 + kbase);
    const v8h* p1 = reinterpret_cast<const v8h*>(mat + m * 64 + kbase + 16);
    return cat16(*p0, *p1);
}

// B operand (K x N = 32x16 f16): lane L: n = nt*16 + (L&15); half-wave selects K offset 0/8;
// elements 0-7 hold K=koff..koff+7, elements 8-15 hold K=16+koff..16+koff+7 (matches A).
// LDS tiles are stored "transposed": row = output column, col = k, pitch P halfs.
static __device__ inline v16h load_b64(const _Float16* __restrict__ tile,
                                       int lane, int nt, int kb, int pitch) {
    int col   = nt * 16 + (lane & 15);
    int kbase = kb * 32 + ((lane >> 4) ? 8 : 0);
    const v8h* p = reinterpret_cast<const v8h*>(tile + col * pitch + kbase);
    return cat16(p[0], p[2]);
}

// ---------------------------------------------------------------------------
// Kernel A: per-h weight precompute. grid = H, block = N(=64), thread = state n.
// ---------------------------------------------------------------------------
__global__ __launch_bounds__(64) void s4d_precompute(
    const float* __restrict__ A_re, const float* __restrict__ A_im,
    const float* __restrict__ Cm,   const float* __restrict__ Dm,
    const float* __restrict__ log_step, char* __restrict__ ws)
{
    __shared__ float kred[64][65];   // [lag][n]
    __shared__ float Ksh[64];

    int h = blockIdx.x;
    int n = threadIdx.x;

    char* base = ws + (size_t)h * WS_STRIDE;
    _Float16* T     = (_Float16*)(base + OFF_T);
    _Float16* WinRe = (_Float16*)(base + OFF_WINRE);
    _Float16* WinIm = (_Float16*)(base + OFF_WINIM);
    _Float16* WoRe  = (_Float16*)(base + OFF_WOUTRE);
    _Float16* WoImN = (_Float16*)(base + OFF_WOUTIMN);
    float*    wc    = (float*)(base + OFF_WC);

    float are = fminf(A_re[h * NDIM + n], -1e-4f);
    float aim = A_im[h * NDIM + n];
    float step = __expf(log_step[h]);
    float dre = step * are, dim = step * aim;
    float e   = __expf(dre);
    float wre = e * cosf(dim);
    float wim = e * sinf(dim);

    // coef = Cc * (w - 1) / A
    float zre = wre - 1.0f, zim = wim;
    float inv = 1.0f / (are * are + aim * aim);
    float qre = (zre * are + zim * aim) * inv;
    float qim = (zim * are - zre * aim) * inv;
    float cre = Cm[(h * NDIM + n) * 2 + 0];
    float cim = Cm[(h * NDIM + n) * 2 + 1];
    float kre = cre * qre - cim * qim;
    float kim = cre * qim + cim * qre;

    // K partials: kred[l][n] = Re(coef * w^l)
    {
        float pr = 1.0f, pi = 0.0f;
        for (int l = 0; l < 64; ++l) {
            kred[l][n] = kre * pr - kim * pi;
            float t = pr * wre - pi * wim;
            pi = pr * wim + pi * wre;
            pr = t;
        }
    }
    __syncthreads();
    {   // thread n reduces lag n across states
        float s = 0.0f;
        for (int k = 0; k < 64; ++k) s += kred[n][k];
        Ksh[n] = s;
    }
    __syncthreads();

    // Toeplitz tile (rows i = within-chunk output pos, cols j = input pos)
    {
        int i = n;
        float dval = Dm[h];
        for (int j = 0; j < 64; ++j) {
            float v = 0.0f;
            if (j < i)       v = Ksh[i - j];
            else if (j == i) v = Ksh[0] + dval;
            T[i * 64 + j] = (_Float16)v;
        }
    }

    // Win[n][j] = w^{63-j}; leaves p = w^64 -> chunk decay
    {
        float pr = 1.0f, pi = 0.0f;
        for (int j = 63; j >= 0; --j) {
            WinRe[n * 64 + j] = (_Float16)pr;
            WinIm[n * 64 + j] = (_Float16)pi;
            float t = pr * wre - pi * wim;
            pi = pr * wim + pi * wre;
            pr = t;
        }
        wc[n]      = pr;   // Re(w^64)
        wc[64 + n] = pi;   // Im(w^64)
    }

    // Wout[i][n] = coef * w^{i+1} (store Re and -Im)
    {
        float pr = wre, pi = wim;
        for (int i = 0; i < 64; ++i) {
            WoRe[i * 64 + n]  = (_Float16)( kre * pr - kim * pi);
            WoImN[i * 64 + n] = (_Float16)(-(kre * pi + kim * pr));
            float t = pr * wre - pi * wim;
            pi = pr * wim + pi * wre;
            pr = t;
        }
    }
}

// ---------------------------------------------------------------------------
// Kernel B: one (b,h) per block, 128 threads = 4 waves; wave w owns m-tile w.
// ---------------------------------------------------------------------------
__global__ __launch_bounds__(128) void s4d_main(
    const float* __restrict__ u, const char* __restrict__ ws,
    float* __restrict__ out)
{
    constexpr int UP = 72;  // f16 B-tile pitch (halfs), 16B-aligned rows
    constexpr int PP = 65;  // f32 P pitch (floats)

    __shared__ __align__(16) _Float16 Ut[64 * UP];    // [q][j]
    __shared__ __align__(16) _Float16 SreT[64 * UP];  // [q][n]
    __shared__ __align__(16) _Float16 SimT[64 * UP];  // [q][n]
    __shared__ float Pre[64 * PP];                    // [q][n]
    __shared__ float Pim[64 * PP];

    int bh = blockIdx.x;                 // bh = b*H + h
    int h  = bh & (HDIM - 1);
    const float* ub = u   + (size_t)bh * LDIM;
    float*       ob = out + (size_t)bh * LDIM;

    const char* base = ws + (size_t)h * WS_STRIDE;
    const _Float16* T     = (const _Float16*)(base + OFF_T);
    const _Float16* WinRe = (const _Float16*)(base + OFF_WINRE);
    const _Float16* WinIm = (const _Float16*)(base + OFF_WINIM);
    const _Float16* WoRe  = (const _Float16*)(base + OFF_WOUTRE);
    const _Float16* WoImN = (const _Float16*)(base + OFF_WOUTIMN);
    const float*    wc    = (const float*)(base + OFF_WC);

    int tid  = threadIdx.x;
    int lane = tid & 31;
    int mt   = tid >> 5;                 // wave id = m-tile

    // stage u chunk-matrix: l = q*64 + j  ->  Ut[q][j] (f16)
    for (int k = 0; k < 32; ++k) {
        int l = tid + k * 128;
        float v = ub[l];
        Ut[(l >> 6) * UP + (l & 63)] = (_Float16)v;
    }
    __syncthreads();

    // P_re/P_im = Win{Re,Im} @ U   (M=n, N=q, K=j)
    for (int part = 0; part < 2; ++part) {
        const _Float16* Amat = part ? WinIm : WinRe;
        float*          Pout = part ? Pim   : Pre;
        for (int nt = 0; nt < 4; ++nt) {
            v8f acc = {};
            for (int kb = 0; kb < 2; ++kb) {
                v16h a  = load_a64(Amat, lane, mt, kb);
                v16h bm = load_b64(Ut, lane, nt, kb, UP);
                acc = __builtin_amdgcn_wmma_f32_16x16x32_f16(
                    false, a, false, bm, (short)0, acc, false, false);
            }
            int col = nt * 16 + (lane & 15);
            int m0  = mt * 16 + ((lane >> 4) << 3);
#pragma unroll
            for (int r = 0; r < 8; ++r) Pout[col * PP + m0 + r] = acc[r];
        }
    }
    __syncthreads();

    // inter-chunk scan: s[q] stored (pre-update), then s <- w^64*s + P[:,q]
    if (tid < 64) {
        int n = tid;
        float wcre = wc[n], wcim = wc[64 + n];
        float sre = 0.0f, sim = 0.0f;
        for (int q = 0; q < 64; ++q) {
            SreT[q * UP + n] = (_Float16)sre;
            SimT[q * UP + n] = (_Float16)sim;
            float pre = Pre[q * PP + n];
            float pim = Pim[q * PP + n];
            float nre = wcre * sre - wcim * sim + pre;
            sim       = wcre * sim + wcim * sre + pim;
            sre       = nre;
        }
    }
    __syncthreads();

    // Y = T@U + WoRe@Sre + WoImN@Sim   (M=i, N=q, K=64)
    for (int nt = 0; nt < 4; ++nt) {
        v8f acc = {};
        for (int kb = 0; kb < 2; ++kb) {
            v16h a  = load_a64(T, lane, mt, kb);
            v16h bm = load_b64(Ut, lane, nt, kb, UP);
            acc = __builtin_amdgcn_wmma_f32_16x16x32_f16(
                false, a, false, bm, (short)0, acc, false, false);
        }
        for (int kb = 0; kb < 2; ++kb) {
            v16h a  = load_a64(WoRe, lane, mt, kb);
            v16h bm = load_b64(SreT, lane, nt, kb, UP);
            acc = __builtin_amdgcn_wmma_f32_16x16x32_f16(
                false, a, false, bm, (short)0, acc, false, false);
        }
        for (int kb = 0; kb < 2; ++kb) {
            v16h a  = load_a64(WoImN, lane, mt, kb);
            v16h bm = load_b64(SimT, lane, nt, kb, UP);
            acc = __builtin_amdgcn_wmma_f32_16x16x32_f16(
                false, a, false, bm, (short)0, acc, false, false);
        }
        int qcol = nt * 16 + (lane & 15);
        int i0   = mt * 16 + ((lane >> 4) << 3);
        float* dst = ob + qcol * 64 + i0;
#pragma unroll
        for (int r = 0; r < 8; ++r) dst[r] = acc[r];
    }
}

// ---------------------------------------------------------------------------
extern "C" void kernel_launch(void* const* d_in, const int* in_sizes, int n_in,
                              void* d_out, int out_size, void* d_ws, size_t ws_size,
                              hipStream_t stream) {
    const float* u        = (const float*)d_in[0];
    const float* A_re     = (const float*)d_in[1];
    const float* A_im     = (const float*)d_in[2];
    const float* Cm       = (const float*)d_in[3];
    const float* Dm       = (const float*)d_in[4];
    const float* log_step = (const float*)d_in[5];

    s4d_precompute<<<HDIM, NDIM, 0, stream>>>(A_re, A_im, Cm, Dm, log_step,
                                              (char*)d_ws);
    s4d_main<<<BDIM * HDIM, 128, 0, stream>>>(u, (const char*)d_ws,
                                              (float*)d_out);
}
